// transformer_model_41145786696166
// MI455X (gfx1250) — matrix operand order repaired and checked by reading, not compiled, not run on hardware
//
#include <hip/hip_runtime.h>
#include <hip/hip_bf16.h>

// ---------------------------------------------------------------------------
// BERT attention on gfx1250 (MI455X): bf16 WMMA (16x16x32) throughout.
//   Kernel 1: QKV projections, fused bias + 1/sqrt(D) (into Q) + head split.
//   Kernel 2: flash attention (online softmax) per (b, h, 128-row F tile).
//     - K tile staged by the Tensor Data Mover (TENSOR_LOAD_TO_LDS, TENSORcnt)
//       with hardware LDS padding; async-LDS fallback if builtin absent.
//     - mask tile staged via GLOBAL_LOAD_ASYNC_TO_LDS_B128 (ASYNCcnt).
// ---------------------------------------------------------------------------

constexpr int BATCH = 2;
constexpr int FDIM  = 2048;     // F
constexpr int TDIM  = 2048;     // T
constexpr int NH    = 16;       // heads
constexpr int HD    = 64;       // size per head
constexpr int WID   = NH * HD;  // 1024
constexpr int MROWS = BATCH * FDIM;  // 4096

typedef __bf16 v2bf  __attribute__((ext_vector_type(2)));
typedef __bf16 v8bf  __attribute__((ext_vector_type(8)));
typedef __bf16 v16bf __attribute__((ext_vector_type(16)));
typedef float  v2f   __attribute__((ext_vector_type(2)));
typedef float  v8f   __attribute__((ext_vector_type(8)));
typedef unsigned int u32x4 __attribute__((ext_vector_type(4)));
typedef int          i32x4 __attribute__((ext_vector_type(4)));
typedef int          i32x8 __attribute__((ext_vector_type(8)));

__device__ __forceinline__ v16bf cat16(v8bf a, v8bf b) {
  return __builtin_shufflevector(a, b, 0,1,2,3,4,5,6,7,8,9,10,11,12,13,14,15);
}

__device__ __forceinline__ v8f wmma_bf16(v16bf a, v16bf b, v8f c) {
  // D = A(16x32 bf16) * B(32x16 bf16) + C(16x16 f32)
  return __builtin_amdgcn_wmma_f32_16x16x32_bf16(
      false, a, false, b, (short)0, c, false, false);
}

// Pack two f32 into one dword of bf16 (lowers to v_cvt_pk_bf16_f32).
__device__ __forceinline__ uint32_t pack_bf16(float a, float b) {
  v2f f = {a, b};
  v2bf h = __builtin_convertvector(f, v2bf);
  return __builtin_bit_cast(uint32_t, h);
}

// 16-lane (half-wave) reductions via ds_swizzle; xor masks 1/2/4/8 stay
// within each group of 16 lanes of the wave32.
#define SWZF(v, pat) __int_as_float(__builtin_amdgcn_ds_swizzle(__float_as_int(v), (pat)))
__device__ __forceinline__ float red_max16(float x) {
  x = fmaxf(x, SWZF(x, 0x041f));
  x = fmaxf(x, SWZF(x, 0x081f));
  x = fmaxf(x, SWZF(x, 0x101f));
  x = fmaxf(x, SWZF(x, 0x201f));
  return x;
}
__device__ __forceinline__ float red_sum16(float x) {
  x += SWZF(x, 0x041f);
  x += SWZF(x, 0x081f);
  x += SWZF(x, 0x101f);
  x += SWZF(x, 0x201f);
  return x;
}

// Async global -> LDS copy, 16B per lane, tracked by ASYNCcnt.
__device__ __forceinline__ void async_copy_b128(
    __attribute__((address_space(3))) void* lds_dst, const void* gsrc) {
  asm volatile("global_load_async_to_lds_b128 %0, %1, off"
               :
               : "v"(lds_dst), "v"(gsrc)
               : "memory");
}
__device__ __forceinline__ void wait_asynccnt0() {
  asm volatile("s_wait_asynccnt 0x0" ::: "memory");
}

// ---- Tensor Data Mover: load a 64x32-element bf16 tile (row-major, row
// length 64, tensor row stride 64) into LDS with 16B pad per 128B row,
// reproducing a [32][72] bf16 LDS layout.  D# built per ISA ch.8.
#if defined(__has_builtin)
#if __has_builtin(__builtin_amdgcn_tensor_load_to_lds)
#define HAVE_TDM 1
#endif
#endif

#ifdef HAVE_TDM
__device__ __forceinline__ void tdm_load_k_tile(uint32_t lds_addr,
                                                const void* gaddr) {
  uint64_t ga = (uint64_t)(uintptr_t)gaddr;
  u32x4 g0;
  g0[0] = 1u;                                   // count=1 (valid user D#)
  g0[1] = lds_addr;                             // LDS byte address
  g0[2] = (uint32_t)(ga & 0xffffffffu);         // global_addr[31:0]
  g0[3] = (uint32_t)((ga >> 32) & 0x01ffffffu)  // global_addr[56:32]
          | (2u << 30);                         // type = 2 ("image")
  i32x8 g1;
  // data_size=1 (2B) @16 | pad_enable @20 | pad_interval=4 (128B) @22
  // | pad_amount=3 (16B) @25 ; workgroup_mask = 0
  g1[0] = (1 << 16) | (1 << 20) | (4 << 22) | (3 << 25);
  g1[1] = (int)(64u << 16);     // tensor_dim0 = 64 (bits 79:48)
  g1[2] = (int)(2048u << 16);   // tensor_dim0 hi=0; tensor_dim1 = 2048 (111:80)
  g1[3] = (int)(64u << 16);     // tensor_dim1 hi=0; tile_dim0 = 64 (127:112)
  g1[4] = 32;                   // tile_dim1 = 32 (143:128); tile_dim2 = 0
  g1[5] = 64;                   // tensor_dim0_stride = 64 (207:160)
  g1[6] = 0;
  g1[7] = 0;
  i32x4 z4 = {0, 0, 0, 0};
#if __clang_major__ >= 23
  i32x8 z8 = {0, 0, 0, 0, 0, 0, 0, 0};
  __builtin_amdgcn_tensor_load_to_lds(g0, g1, z4, z4, z8, 0);
#else
  __builtin_amdgcn_tensor_load_to_lds(g0, g1, z4, z4, 0);
#endif
}
#endif

// ---------------------------------------------------------------------------
// Kernel 1: QKV projection. blockIdx.z in {0,1,2} -> Q, K, V.
// C[m,n] = X[m,:] @ W[:,n] + bias[n]; Q additionally scaled by 0.125.
// Output written bf16 in [b, h, s, d] layout.
// ---------------------------------------------------------------------------
constexpr int PBM = 64, PBN = 64, PBK = 32;

__global__ __launch_bounds__(256) void qkv_proj_kernel(
    const float* __restrict__ from_t, const float* __restrict__ to_t,
    const float* __restrict__ Wq, const float* __restrict__ Wk,
    const float* __restrict__ Wv, const float* __restrict__ bq,
    const float* __restrict__ bk, const float* __restrict__ bv,
    __bf16* __restrict__ Qw, __bf16* __restrict__ Kw, __bf16* __restrict__ Vw)
{
  const int p = blockIdx.z;
  const float* X    = (p == 0) ? from_t : to_t;
  const float* Wm   = (p == 0) ? Wq : (p == 1) ? Wk : Wv;
  const float* bias = (p == 0) ? bq : (p == 1) ? bk : bv;
  __bf16* out       = (p == 0) ? Qw : (p == 1) ? Kw : Vw;
  const float oscale = (p == 0) ? 0.125f : 1.0f;   // fold 1/sqrt(64) into Q

  const int m0 = blockIdx.x * PBM;
  const int n0 = blockIdx.y * PBN;

  __shared__ __attribute__((aligned(16))) __bf16 As[PBM][PBK + 8];  // [m][k]
  __shared__ __attribute__((aligned(16))) __bf16 Bs[PBN][PBK + 8];  // [n][k]

  const int tid  = threadIdx.x;
  const int lane = tid & 31;
  const int wave = tid >> 5;
  const int mw   = wave & 3;   // 4 m sub-tiles of 16
  const int nw   = wave >> 2;  // 2 n sub-tiles of 32

  v8f acc0 = {};
  v8f acc1 = {};

  for (int k0 = 0; k0 < WID; k0 += PBK) {
    __syncthreads();
    // Stage A: 64x32 f32 -> packed bf16 dwords (coalesced float2 reads).
    #pragma unroll
    for (int i = 0; i < 4; ++i) {
      int idx = tid + i * 256;            // 1024 f32 pairs
      int r = idx >> 4, c2 = idx & 15;    // row, pair column
      const float* xp = X + (size_t)(m0 + r) * WID + k0 + 2 * c2;
      float2 xv = *(const float2*)xp;
      *(uint32_t*)&As[r][2 * c2] = pack_bf16(xv.x, xv.y);
    }
    // Stage B transposed: W[k0+2rp..+1, n0+c] -> Bs[c][2rp..+1] packed.
    #pragma unroll
    for (int i = 0; i < 4; ++i) {
      int idx = tid + i * 256;            // 1024 k-pairs
      int rp = idx >> 6, c = idx & 63;    // k-pair, n column
      const float* wp = Wm + (size_t)(k0 + 2 * rp) * WID + n0 + c;
      float w0 = wp[0];
      float w1 = wp[WID];
      *(uint32_t*)&Bs[c][2 * rp] = pack_bf16(w0, w1);
    }
    __syncthreads();

    // A fragment: M = lane%16, K runs [lo..lo+15] contiguous.
    const int arow = 16 * mw + (lane & 15);
    const int lo   = (lane >> 4) << 4;
    v16bf afrag = cat16(*(const v8bf*)&As[arow][lo],
                        *(const v8bf*)&As[arow][lo + 8]);

    // B fragments: N = lane%16, K = h + 16*(lane>=16) -> contiguous run.
    const int kk  = (lane >> 4) << 4;
    const int br0 = 32 * nw + (lane & 15);
    v16bf bf0 = cat16(*(const v8bf*)&Bs[br0][kk],
                      *(const v8bf*)&Bs[br0][kk + 8]);
    v16bf bf1 = cat16(*(const v8bf*)&Bs[br0 + 16][kk],
                      *(const v8bf*)&Bs[br0 + 16][kk + 8]);

    acc0 = wmma_bf16(afrag, bf0, acc0);
    acc1 = wmma_bf16(afrag, bf1, acc1);
  }

  // Epilogue: bias, scale, convert to bf16, write [b,h,s,d]
  const int nA = n0 + 32 * nw + (lane & 15);
  const int nB = nA + 16;
  const float biasA = bias[nA];
  const float biasB = bias[nB];
  const int hA = nA >> 6, dA = nA & 63;
  const int hB = nB >> 6, dB = nB & 63;
  const int mbase = m0 + 16 * mw + ((lane >> 4) << 3);   // M = r + 8*(lane>=16)
  #pragma unroll
  for (int r = 0; r < 8; ++r) {
    int m = mbase + r;
    int b = m >> 11;        // / 2048
    int s = m & 2047;
    size_t baseA = ((((size_t)b * NH + hA) * FDIM) + s) * HD + dA;
    size_t baseB = ((((size_t)b * NH + hB) * FDIM) + s) * HD + dB;
    out[baseA] = (__bf16)((acc0[r] + biasA) * oscale);
    out[baseB] = (__bf16)((acc1[r] + biasB) * oscale);
  }
}

// ---------------------------------------------------------------------------
// Kernel 2: flash attention. Block = 256 threads (8 waves), each wave owns a
// 16-row Q tile; block covers 128 F rows of one (b, h). Streams K/V/mask in
// 32-col tiles through LDS with online softmax.
// ---------------------------------------------------------------------------
__global__ __launch_bounds__(256) void attention_kernel(
    const __bf16* __restrict__ Qw, const __bf16* __restrict__ Kw,
    const __bf16* __restrict__ Vw, const int* __restrict__ mask,
    float* __restrict__ outp)
{
  const int b  = blockIdx.z;
  const int h  = blockIdx.y;
  const int f0 = blockIdx.x * 128;

  const int tid  = threadIdx.x;
  const int lane = tid & 31;
  const int wave = tid >> 5;

  __shared__ __attribute__((aligned(16))) __bf16   Ks[32][HD + 8];   // [t'][d]
  __shared__ __attribute__((aligned(16))) __bf16   Vs[HD][32 + 8];   // [d][t']
  __shared__ __attribute__((aligned(16))) uint32_t Ps[8][16][20];    // dword j = (t'=j, t'=j+16)
  __shared__ __attribute__((aligned(16))) int      Ms[128][32];      // mask tile

  const size_t head = (((size_t)b * NH + h) * FDIM) * HD;
  const __bf16* Qh = Qw + head;
  const __bf16* Kh = Kw + head;
  const __bf16* Vh = Vw + head;

  const int ncol     = lane & 15;
  const int lo       = (lane >> 4) << 3;    // A frag K-run base: 0 or 8
  const int kk       = (lane >> 4) << 3;    // B frag K-run base: 0 or 8
  const int halfBase = (lane >> 4) << 3;    // C frag row base: 0 or 8
  const int mrow     = f0 + 16 * wave + ncol;

  // Q fragments for d=[0..31] and d=[32..63], held in registers (scaled by 1/8)
  v16bf qa[2];
  #pragma unroll
  for (int j = 0; j < 2; ++j) {
    const __bf16* qp = Qh + (size_t)mrow * HD + 32 * j;
    qa[j] = cat16(*(const v8bf*)(qp + lo), *(const v8bf*)(qp + lo + 16));
  }

  v8f o0 = {}, o1 = {}, o2 = {}, o3 = {};
  float rmax[8], rsum[8];
  #pragma unroll
  for (int r = 0; r < 8; ++r) { rmax[r] = -1e30f; rsum[r] = 0.0f; }

#ifdef HAVE_TDM
  const uint32_t ks_lds =
      (uint32_t)(uintptr_t)(__attribute__((address_space(3))) void*)&Ks[0][0];
#else
  const int krow_st = wave * 4 + (lane >> 3);   // 0..31
  const int kcol_st = (lane & 7) * 8;           // bf16 offset, 16B chunks
#endif

  for (int t0 = 0; t0 < TDIM; t0 += 32) {
    __syncthreads();

    // --- K tile: Tensor Data Mover (TENSORcnt) with hardware LDS pad. ---
#ifdef HAVE_TDM
    if (tid < 32) {
      tdm_load_k_tile(ks_lds, Kh + (size_t)t0 * HD);
      __builtin_amdgcn_s_wait_tensorcnt(0);
    }
#else
    {
      const __bf16* gsrc = Kh + (size_t)(t0 + krow_st) * HD + kcol_st;
      async_copy_b128(
          (__attribute__((address_space(3))) void*)&Ks[krow_st][kcol_st], gsrc);
    }
#endif

    // --- mask tile: async global -> LDS (ASYNCcnt), 4 rows per b128. ---
    #pragma unroll
    for (int i = 0; i < 4; ++i) {
      int row = 16 * wave + 4 * i + (lane >> 3);   // 0..127
      const int* gm = mask + ((size_t)b * FDIM + (f0 + row)) * TDIM + t0
                      + (lane & 7) * 4;
      async_copy_b128(
          (__attribute__((address_space(3))) void*)&Ms[row][(lane & 7) * 4], gm);
    }

    // --- V tile: transposed staging with packed bf16 dword stores. ---
    const uint32_t* Vg = (const uint32_t*)(Vh + (size_t)t0 * HD);
    #pragma unroll
    for (int i = 0; i < 2; ++i) {
      int idx = tid + i * 256;          // 512 2x2 blocks
      int rp = idx >> 5;                // t' pair 0..15
      int c  = idx & 31;                // d pair
      uint32_t a = Vg[(2 * rp) * 32 + c];
      uint32_t bdw = Vg[(2 * rp + 1) * 32 + c];
      uint32_t dlo = (a & 0xffffu) | (bdw << 16);          // d=2c : t'=2rp,2rp+1
      uint32_t dhi = (a >> 16) | (bdw & 0xffff0000u);      // d=2c+1
      *(uint32_t*)&Vs[2 * c][2 * rp]     = dlo;
      *(uint32_t*)&Vs[2 * c + 1][2 * rp] = dhi;
    }
    if (t0 + 32 < TDIM) {               // global_prefetch_b8 next tiles
      const char* nk = (const char*)(Kh + (size_t)(t0 + 32) * HD);
      const char* nv = (const char*)(Vg + 1024);
      __builtin_prefetch(nk + tid * 16, 0, 1);
      __builtin_prefetch(nv + tid * 16, 0, 1);
    }
    wait_asynccnt0();
    __syncthreads();

    // Scores: S(16x32) = Q(16x64) @ K^T, two 16x16 WMMA column tiles.
    v8f sc[2];
    #pragma unroll
    for (int c = 0; c < 2; ++c) {
      int krow = (c << 4) + ncol;       // t' row in Ks
      v16bf kb0 = cat16(*(const v8bf*)&Ks[krow][kk],
                        *(const v8bf*)&Ks[krow][kk + 16]);
      v16bf kb1 = cat16(*(const v8bf*)&Ks[krow][32 + kk],
                        *(const v8bf*)&Ks[krow][32 + kk + 16]);
      v8f s = {};
      s = wmma_bf16(qa[0], kb0, s);
      s = wmma_bf16(qa[1], kb1, s);
      sc[c] = s;
    }

    // Additive BERT mask from LDS: (1 - mask) * -10000
    #pragma unroll
    for (int r = 0; r < 8; ++r) {
      int mrl = 16 * wave + halfBase + r;
      sc[0][r] += -10000.0f * (1.0f - (float)Ms[mrl][ncol]);
      sc[1][r] += -10000.0f * (1.0f - (float)Ms[mrl][16 + ncol]);
    }

    // Online softmax update (per-row stats replicated across 16-lane halves)
    #pragma unroll
    for (int r = 0; r < 8; ++r) {
      float tm   = red_max16(fmaxf(sc[0][r], sc[1][r]));
      float nm   = fmaxf(rmax[r], tm);
      float corr = __expf(rmax[r] - nm);
      float p0   = __expf(sc[0][r] - nm);
      float p1   = __expf(sc[1][r] - nm);
      sc[0][r] = p0;
      sc[1][r] = p1;
      rsum[r] = rsum[r] * corr + red_sum16(p0 + p1);
      rmax[r] = nm;
      o0[r] *= corr; o1[r] *= corr; o2[r] *= corr; o3[r] *= corr;
    }

    // P: C-layout -> A-layout via per-wave LDS staging; dword j of row m
    // holds the bf16 pair (t'=j, t'=j+16), one packed b32 store per row.
    #pragma unroll
    for (int r = 0; r < 8; ++r) {
      Ps[wave][halfBase + r][ncol] = pack_bf16(sc[0][r], sc[1][r]);
    }
    // A-frag read: dwords lo..lo+7 of row m = lane%16, then deinterleave.
    v16bf praw = *(const v16bf*)&Ps[wave][ncol][lo];
    v16bf pa = __builtin_shufflevector(praw, praw,
        0,2,4,6,8,10,12,14, 1,3,5,7,9,11,13,15);

    // ctx += P(16x32) @ V(32x64): 4 WMMAs over d chunks of 16.
    v16bf vb0 = cat16(*(const v8bf*)&Vs[ncol][kk],      *(const v8bf*)&Vs[ncol][kk + 16]);
    v16bf vb1 = cat16(*(const v8bf*)&Vs[16 + ncol][kk], *(const v8bf*)&Vs[16 + ncol][kk + 16]);
    v16bf vb2 = cat16(*(const v8bf*)&Vs[32 + ncol][kk], *(const v8bf*)&Vs[32 + ncol][kk + 16]);
    v16bf vb3 = cat16(*(const v8bf*)&Vs[48 + ncol][kk], *(const v8bf*)&Vs[48 + ncol][kk + 16]);
    o0 = wmma_bf16(pa, vb0, o0);
    o1 = wmma_bf16(pa, vb1, o1);
    o2 = wmma_bf16(pa, vb2, o2);
    o3 = wmma_bf16(pa, vb3, o3);
  }

  // Normalize and write ctx as f32 in [B, F, H*D].
  #pragma unroll
  for (int r = 0; r < 8; ++r) {
    float inv = (rsum[r] > 0.0f) ? (1.0f / rsum[r]) : 0.0f;
    int f = f0 + 16 * wave + halfBase + r;
    float* orow = outp + ((size_t)b * FDIM + f) * WID + h * HD;
    orow[0  + ncol] = o0[r] * inv;
    orow[16 + ncol] = o1[r] * inv;
    orow[32 + ncol] = o2[r] * inv;
    orow[48 + ncol] = o3[r] * inv;
  }
}

// ---------------------------------------------------------------------------
extern "C" void kernel_launch(void* const* d_in, const int* in_sizes, int n_in,
                              void* d_out, int out_size, void* d_ws, size_t ws_size,
                              hipStream_t stream) {
  const float* from_t = (const float*)d_in[0];
  const float* to_t   = (const float*)d_in[1];
  const float* Wq     = (const float*)d_in[2];
  const float* bq     = (const float*)d_in[3];
  const float* Wk     = (const float*)d_in[4];
  const float* bk     = (const float*)d_in[5];
  const float* Wv     = (const float*)d_in[6];
  const float* bv     = (const float*)d_in[7];
  const int*   mask   = (const int*)d_in[8];
  float* out = (float*)d_out;

  const size_t headElems = (size_t)BATCH * NH * FDIM * HD;  // 4 Mi elems each
  __bf16* Qw = (__bf16*)d_ws;
  __bf16* Kw = Qw + headElems;
  __bf16* Vw = Kw + headElems;

  dim3 gridP(MROWS / PBM, WID / PBN, 3);   // 64 x 16 x 3
  qkv_proj_kernel<<<gridP, 256, 0, stream>>>(from_t, to_t, Wq, Wk, Wv,
                                             bq, bk, bv, Qw, Kw, Vw);

  dim3 gridA(FDIM / 128, NH, BATCH);       // 16 x 16 x 2
  attention_kernel<<<gridA, 256, 0, stream>>>(Qw, Kw, Vw, mask, out);
}
